// DynamicFeatureUpdateHead_74363063762954
// MI455X (gfx1250) — matrix operand order repaired and checked by reading, not compiled, not run on hardware
//
#include <hip/hip_runtime.h>

typedef __attribute__((ext_vector_type(16))) _Float16 v16h;
typedef __attribute__((ext_vector_type(8)))  float    v8f;

#define HD   128   // head dim
#define NSP  64    // spatial positions
#define CIN  256   // input channels
#define SCALE 0.08838834764831843f  // 1/sqrt(128)

// ---------------- WMMA helpers (CDNA5 16x16x32 f16 -> f32) ----------------

__device__ inline v8f wmma16(v16h a, v16h b, v8f c) {
  return __builtin_amdgcn_wmma_f32_16x16x32_f16(
      /*neg_a=*/false, a, /*neg_b=*/false, b,
      /*c_mod=*/(short)0, c, /*reuse_a=*/false, /*reuse_b=*/false);
}

// A fragment: 16x32 tile of row-major [M][K] f16 matrix, leading dim lda.
// Lane L<16: row m0+L, K = k0+{0..7, 16..23}; lane>=16: K = k0+{8..15, 24..31}.
// Both 8-half groups are contiguous -> lowers to 2x b128 loads.
__device__ inline v16h load_a16(const _Float16* A, int lda, int m0, int k0) {
  const int lane = threadIdx.x & 31;
  const _Float16* p = A + (m0 + (lane & 15)) * lda + k0 + ((lane >> 4) << 3);
  v16h a;
#pragma unroll
  for (int i = 0; i < 8; ++i) { a[i] = p[i]; a[8 + i] = p[16 + i]; }
  return a;
}

// Fragment-packed B: element (k,n) of a KxN B-matrix lives at
//   (((k>>5)*ntilesN + (n>>4))*32 + ((k>>3)&1)*16 + (n&15))*16 + ((k>>4)&1)*8 + (k&7)
// so each lane's 16 halves for tile (kk,nt) are one contiguous v16h (32B).
__device__ inline int pack_pos(int k, int n, int ntilesN) {
  return (((k >> 5) * ntilesN + (n >> 4)) * 32 + (((k >> 3) & 1) << 4) + (n & 15)) * 16 + (((k >> 4) & 1) << 3) + (k & 7);
}

__device__ inline v16h load_bp(const _Float16* P, int ntilesN, int kk, int nt) {
  const v16h* p = (const v16h*)P;
  return p[(kk * ntilesN + nt) * 32 + (threadIdx.x & 31)];
}

// ---------------- prep kernels ----------------

// plain f32 -> f16 row-major copy (A-side weights)
__global__ void cvt_copy_kernel(const float* __restrict__ s, _Float16* __restrict__ d, int n) {
  int i = blockIdx.x * blockDim.x + threadIdx.x;
  if (i < n) d[i] = (_Float16)s[i];
}

// Build fragment-packed B (KxN) from f32 source.
// val(k,n) = transposeSrc ? s[n*srcLd + colOff + k] : s[k*srcLd + colOff + n]
__global__ void pack_b_kernel(const float* __restrict__ s, _Float16* __restrict__ d,
                              int K, int N, int srcLd, int colOff, int transposeSrc) {
  int i = blockIdx.x * blockDim.x + threadIdx.x;
  if (i >= K * N) return;
  int k = i / N, n = i - k * N;
  float v = transposeSrc ? s[n * srcLd + colOff + k] : s[k * srcLd + colOff + n];
  d[pack_pos(k, n, N >> 4)] = (_Float16)v;
}

// Pack x (B batches of [256][64]) into per-batch fragment-packed f16.
__global__ void pack_x_kernel(const float* __restrict__ x, _Float16* __restrict__ xp) {
  int i = blockIdx.x * blockDim.x + threadIdx.x;
  if (i >= 64 * CIN * NSP) return;
  int b = i >> 14;            // /16384
  int r = i & 16383;
  int k = r >> 6, n = r & 63; // channel, spatial
  xp[b * (CIN * NSP) + pack_pos(k, n, 4)] = (_Float16)x[i];
}

// ---------------- phase 1: features, move logits, src_p/dst_p ----------------

__global__ __launch_bounds__(256) void phase1_kernel(
    const _Float16* __restrict__ xp,     // [B] fragment-packed (256x64)
    const _Float16* __restrict__ wsrc_h, // [128][256] f16 row-major (A)
    const _Float16* __restrict__ wdst_h,
    const _Float16* __restrict__ warr_h,
    const float* __restrict__ b_src,
    const float* __restrict__ b_dst,
    const float* __restrict__ b_arr,
    const _Float16* __restrict__ w1at_p, // packed B: (128x256) = w1[:, :128]^T
    const _Float16* __restrict__ w1bt_p, // packed B: (128x256) = w1[:, 128:]^T
    _Float16* __restrict__ farr_p,       // [B] packed B: (128x64)
    float* __restrict__ src_p,           // [B][64][256]
    float* __restrict__ dst_p,           // [B][64][256]
    float* __restrict__ move_out)        // [B][64][64]
{
  __shared__ _Float16 fsrcT[NSP * HD];                       // [n][d] row-major (A)
  __shared__ _Float16 fdstT[NSP * HD];                       // [m][d] row-major (A)
  __shared__ __align__(32) _Float16 fdstP[HD * NSP];         // packed B (128x64)

  const int b      = blockIdx.x;
  const int wv     = threadIdx.x >> 5;
  const int lane   = threadIdx.x & 31;
  const int lane15 = lane & 15;
  const int rowoff = (lane >> 4) << 3;

  const _Float16* xpb = xp + b * (CIN * NSP);

  // f_W = W(128x256) @ xb(256x64) + bias ; wave wv owns m-tile wv, all 4 n-tiles
  for (int which = 0; which < 3; ++which) {
    const _Float16* W  = (which == 0) ? wsrc_h : (which == 1) ? wdst_h : warr_h;
    const float* bias  = (which == 0) ? b_src  : (which == 1) ? b_dst  : b_arr;
    v8f acc[4] = {};
    for (int kk = 0; kk < 8; ++kk) {
      v16h a = load_a16(W, CIN, wv * 16, kk * 32);
#pragma unroll
      for (int j = 0; j < 4; ++j) {
        v16h bb = load_bp(xpb, 4, kk, j);
        acc[j] = wmma16(a, bb, acc[j]);
      }
    }
#pragma unroll
    for (int j = 0; j < 4; ++j) {
      int n = j * 16 + lane15;
#pragma unroll
      for (int i = 0; i < 8; ++i) {
        int m = wv * 16 + rowoff + i;
        _Float16 h = (_Float16)(acc[j][i] + bias[m]);
        if (which == 0)      { fsrcT[n * HD + m] = h; }
        else if (which == 1) { fdstT[n * HD + m] = h; fdstP[pack_pos(m, n, 4)] = h; }
        else                 { farr_p[b * (HD * NSP) + pack_pos(m, n, 4)] = h; }
      }
    }
  }
  __syncthreads();

  // move_logits = fsrcT(64x128) @ f_dst(128x64) * scale
  {
    const int mt  = wv & 3;
    const int ntb = (wv >> 2) * 2;
    for (int j = 0; j < 2; ++j) {
      const int nt = ntb + j;
      v8f acc = {};
      for (int kk = 0; kk < 4; ++kk) {
        v16h a  = load_a16(fsrcT, HD, mt * 16, kk * 32);
        v16h bb = load_bp(fdstP, 4, kk, nt);
        acc = wmma16(a, bb, acc);
      }
      const int ncol = nt * 16 + lane15;
#pragma unroll
      for (int i = 0; i < 8; ++i) {
        int nsrc = mt * 16 + rowoff + i;
        move_out[b * (NSP * NSP) + nsrc * NSP + ncol] = acc[i] * SCALE;
      }
    }
  }

  // src_p = fsrcT(64x128) @ w1a^T(128x256); dst_p = fdstT @ w1b^T
  for (int which = 0; which < 2; ++which) {
    const _Float16* A  = which ? fdstT  : fsrcT;
    const _Float16* Bm = which ? w1bt_p : w1at_p;
    float* dst         = which ? dst_p  : src_p;
    const int mt  = wv & 3;
    const int ntb = (wv >> 2) * 8;
    for (int j = 0; j < 8; ++j) {
      const int nt = ntb + j;
      v8f acc = {};
      for (int kk = 0; kk < 4; ++kk) {
        v16h a  = load_a16(A, HD, mt * 16, kk * 32);
        v16h bb = load_bp(Bm, 16, kk, nt);
        acc = wmma16(a, bb, acc);
      }
      const int h = nt * 16 + lane15;
#pragma unroll
      for (int i = 0; i < 8; ++i) {
        int r = mt * 16 + rowoff + i;
        dst[(b * NSP + r) * (2 * HD) + h] = acc[i];
      }
    }
  }
}

// ---------------- phase 2: fused per-(b,n) MLP + arrow logits ----------------

__global__ __launch_bounds__(256) void phase2_kernel(
    const float* __restrict__ src_p,
    const float* __restrict__ dst_p,
    const float* __restrict__ b1,
    const _Float16* __restrict__ w2t_p,  // packed B (256x128) = w2^T
    const float* __restrict__ b2,
    const _Float16* __restrict__ w3t_p,  // packed B (128x128) = w3^T
    const float* __restrict__ b3,
    const _Float16* __restrict__ farr_p, // [B] packed B (128x64)
    float* __restrict__ arrow_out)       // [B][N*N][64]
{
  __shared__ _Float16 h1h[NSP * 2 * HD]; // 64x256 row-major (A)
  __shared__ _Float16 h2h[NSP * HD];     // 64x128 row-major (A)
  __shared__ _Float16 qh [NSP * HD];     // 64x128 row-major (A)

  const int b      = blockIdx.x >> 6;
  const int n      = blockIdx.x & 63;
  const int tid    = threadIdx.x;
  const int wv     = tid >> 5;
  const int lane   = tid & 31;
  const int lane15 = lane & 15;
  const int rowoff = (lane >> 4) << 3;

  // Step 1: h1[m][h] = relu(src_p[b,n,h] + dst_p[b,m,h] + b1[h]) ; thread owns column h
  {
    const int h = tid;
    const float sv = src_p[(b * NSP + n) * 256 + h];
    const float bv = b1[h];
    const float* dp = dst_p + b * NSP * 256 + h;
#pragma unroll 4
    for (int m = 0; m < NSP; ++m) {
      float v = sv + dp[m * 256] + bv;
      h1h[m * 256 + h] = (_Float16)(v > 0.f ? v : 0.f);
    }
  }
  __syncthreads();

  // Step 2: h2 = relu(h1(64x256) @ w2^T(256x128) + b2)
  {
    const int mt  = wv & 3;
    const int ntb = (wv >> 2) * 4;
    v8f acc[4] = {};
    for (int kk = 0; kk < 8; ++kk) {
      v16h a = load_a16(h1h, 256, mt * 16, kk * 32);
#pragma unroll
      for (int j = 0; j < 4; ++j) {
        v16h bb = load_bp(w2t_p, 8, kk, ntb + j);
        acc[j] = wmma16(a, bb, acc[j]);
      }
    }
#pragma unroll
    for (int j = 0; j < 4; ++j) {
      const int o = (ntb + j) * 16 + lane15;
      const float bias = b2[o];
#pragma unroll
      for (int i = 0; i < 8; ++i) {
        float v = acc[j][i] + bias;
        h2h[(mt * 16 + rowoff + i) * HD + o] = (_Float16)(v > 0.f ? v : 0.f);
      }
    }
  }
  __syncthreads();

  // Step 3: q = h2(64x128) @ w3^T(128x128) + b3
  {
    const int mt  = wv & 3;
    const int ntb = (wv >> 2) * 4;
    v8f acc[4] = {};
    for (int kk = 0; kk < 4; ++kk) {
      v16h a = load_a16(h2h, HD, mt * 16, kk * 32);
#pragma unroll
      for (int j = 0; j < 4; ++j) {
        v16h bb = load_bp(w3t_p, 8, kk, ntb + j);
        acc[j] = wmma16(a, bb, acc[j]);
      }
    }
#pragma unroll
    for (int j = 0; j < 4; ++j) {
      const int p = (ntb + j) * 16 + lane15;
      const float bias = b3[p];
#pragma unroll
      for (int i = 0; i < 8; ++i)
        qh[(mt * 16 + rowoff + i) * HD + p] = (_Float16)(acc[j][i] + bias);
    }
  }
  __syncthreads();

  // Step 4: arrow[b, n*64+m, k] = (q(64x128) @ f_arr[b](128x64))[m][k] * scale
  {
    const _Float16* fb = farr_p + b * (HD * NSP);
    float* outb = arrow_out + (size_t)(b * NSP + n) * (NSP * NSP);
    const int mt  = wv & 3;
    const int ntb = (wv >> 2) * 2;
    for (int j = 0; j < 2; ++j) {
      const int nt = ntb + j;
      v8f acc = {};
      for (int kk = 0; kk < 4; ++kk) {
        v16h a  = load_a16(qh, HD, mt * 16, kk * 32);
        v16h bb = load_bp(fb, 4, kk, nt);
        acc = wmma16(a, bb, acc);
      }
      const int k = nt * 16 + lane15;
#pragma unroll
      for (int i = 0; i < 8; ++i) {
        int m = mt * 16 + rowoff + i;
        outb[m * NSP + k] = acc[i] * SCALE;
      }
    }
  }
}

// ---------------- launch ----------------

extern "C" void kernel_launch(void* const* d_in, const int* in_sizes, int n_in,
                              void* d_out, int out_size, void* d_ws, size_t ws_size,
                              hipStream_t stream) {
  (void)in_sizes; (void)n_in; (void)out_size; (void)ws_size;
  const float* x     = (const float*)d_in[0];
  const float* w_src = (const float*)d_in[1];
  const float* b_src = (const float*)d_in[2];
  const float* w_dst = (const float*)d_in[3];
  const float* b_dst = (const float*)d_in[4];
  const float* w_arr = (const float*)d_in[5];
  const float* b_arr = (const float*)d_in[6];
  const float* w1    = (const float*)d_in[7];
  const float* b1    = (const float*)d_in[8];
  const float* w2    = (const float*)d_in[9];
  const float* b2    = (const float*)d_in[10];
  const float* w3    = (const float*)d_in[11];
  const float* b3    = (const float*)d_in[12];

  float* out       = (float*)d_out;
  float* move_out  = out;                       // 64*64*64
  float* arrow_out = out + 64 * 64 * 64;        // 64*4096*64

  char* ws = (char*)d_ws;
  size_t off = 0;
  _Float16* wsrc_h = (_Float16*)(ws + off); off += 128 * 256 * 2;
  _Float16* wdst_h = (_Float16*)(ws + off); off += 128 * 256 * 2;
  _Float16* warr_h = (_Float16*)(ws + off); off += 128 * 256 * 2;
  _Float16* w1at_p = (_Float16*)(ws + off); off += 128 * 256 * 2;
  _Float16* w1bt_p = (_Float16*)(ws + off); off += 128 * 256 * 2;
  _Float16* w2t_p  = (_Float16*)(ws + off); off += 256 * 128 * 2;
  _Float16* w3t_p  = (_Float16*)(ws + off); off += 128 * 128 * 2;
  _Float16* farr_p = (_Float16*)(ws + off); off += (size_t)64 * 128 * 64 * 2;
  _Float16* x_p    = (_Float16*)(ws + off); off += (size_t)64 * 256 * 64 * 2;
  float*    src_p  = (float*)(ws + off);    off += (size_t)64 * 64 * 256 * 4;
  float*    dst_p  = (float*)(ws + off);    off += (size_t)64 * 64 * 256 * 4;

  // A-side weights: plain f16 row-major
  cvt_copy_kernel<<<128, 256, 0, stream>>>(w_src, wsrc_h, 32768);
  cvt_copy_kernel<<<128, 256, 0, stream>>>(w_dst, wdst_h, 32768);
  cvt_copy_kernel<<<128, 256, 0, stream>>>(w_arr, warr_h, 32768);
  // B-side weights: fragment-packed f16
  pack_b_kernel<<<128, 256, 0, stream>>>(w1, w1at_p, 128, 256, 256, 0,   1);
  pack_b_kernel<<<128, 256, 0, stream>>>(w1, w1bt_p, 128, 256, 256, 128, 1);
  pack_b_kernel<<<128, 256, 0, stream>>>(w2, w2t_p,  256, 128, 256, 0,   1);
  pack_b_kernel<<<64,  256, 0, stream>>>(w3, w3t_p,  128, 128, 128, 0,   1);
  pack_x_kernel<<<4096, 256, 0, stream>>>(x, x_p);

  phase1_kernel<<<64, 256, 0, stream>>>(x_p, wsrc_h, wdst_h, warr_h,
                                        b_src, b_dst, b_arr,
                                        w1at_p, w1bt_p, farr_p, src_p, dst_p, move_out);
  phase2_kernel<<<64 * 64, 256, 0, stream>>>(src_p, dst_p, b1, w2t_p, b2, w3t_p, b3,
                                             farr_p, arrow_out);
}
